// CausalSelfAttention_3719441679139
// MI455X (gfx1250) — matrix operand order repaired and checked by reading, not compiled, not run on hardware
//
#include <hip/hip_runtime.h>
#include <hip/hip_bf16.h>

// CDNA5 / gfx1250 causal self-attention:
//   GEMM (bf16 WMMA, f32 accum, double-buffered LDS + prefetch)
//   -> flash attention (WMMA QK^T + PV, online softmax)
//   -> GEMM projection.
// B=2, T=2048, C=1024, H=16, hd=64.

typedef __attribute__((ext_vector_type(16))) __bf16 v16bf;
typedef __attribute__((ext_vector_type(8)))  float v8f;

union Frag {
    v16bf v;
    uint4 q[2];
};

__device__ inline unsigned short f32_bf16(float f) {
    unsigned int u = __float_as_uint(f);
    u += 0x7FFFu + ((u >> 16) & 1u);   // round-to-nearest-even
    return (unsigned short)(u >> 16);
}

__device__ inline v8f zero8() {
    v8f z;
#pragma unroll
    for (int e = 0; e < 8; ++e) z[e] = 0.0f;
    return z;
}

// ---------------------------------------------------------------------------
// GEMM: C[M,N] = A[M,K] (f32) * Bw[K,N] (f32) + bias, via bf16 WMMA.
// mode 0: store f32 to outF (row-major MxN).
// mode 1: scatter QKV: N = 3C; q/k -> [B*H, T, 64] bf16, v -> [B*H, 64, T] bf16.
// Block: 256 threads (8 waves), tile 128x128, K-step 32, double-buffered LDS.
// ---------------------------------------------------------------------------
#define TM 128
#define TN 128
#define TK 32
#define APAD 16
#define BPAD 16

__global__ __launch_bounds__(256)
void gemm_bf16_wmma(const float* __restrict__ A, const float* __restrict__ Bw,
                    const float* __restrict__ bias,
                    int M, int N, int K, int mode,
                    float* __restrict__ outF,
                    unsigned short* __restrict__ qbuf,
                    unsigned short* __restrict__ kbuf,
                    unsigned short* __restrict__ vbuf)
{
    __shared__ __align__(16) unsigned short As[2][TM][TK + APAD];  // 2 x 128 x 48 bf16
    __shared__ __align__(16) unsigned short Bs[2][TK][TN + BPAD];  // 2 x  32 x 144 bf16

    const int tid  = threadIdx.x;
    const int lane = tid & 31;
    const int wave = tid >> 5;     // 0..7
    const int wm   = wave >> 1;    // 0..3 (M direction)
    const int wn   = wave & 1;     // 0..1 (N direction)

    const int row0 = blockIdx.y * TM;
    const int col0 = blockIdx.x * TN;

    const int half  = lane >> 4;    // 0/1
    const int nlane = lane & 15;

    // per-thread staging coordinates
    const int ar = tid >> 1;            // A slab: row 0..127
    const int ac = (tid & 1) * 16;      //          col base 0/16
    const int br = tid >> 3;            // B slab: row 0..31
    const int bc = (tid & 7) * 16;      //          col base 0..112

    v8f acc[2][4];
#pragma unroll
    for (int i = 0; i < 2; ++i)
#pragma unroll
        for (int j = 0; j < 4; ++j) acc[i][j] = zero8();

    // stage one K-slab (f32 global -> bf16 LDS) into buffer `buf`
    auto stage = [&](int buf, int k0) {
#pragma unroll
        for (int cc = 0; cc < 4; ++cc) {
            const int c = ac + cc * 4;
            const float4 f = *(const float4*)&A[(size_t)(row0 + ar) * K + k0 + c];
            *(unsigned int*)&As[buf][ar][c] =
                (unsigned int)f32_bf16(f.x) | ((unsigned int)f32_bf16(f.y) << 16);
            *(unsigned int*)&As[buf][ar][c + 2] =
                (unsigned int)f32_bf16(f.z) | ((unsigned int)f32_bf16(f.w) << 16);
        }
#pragma unroll
        for (int cc = 0; cc < 4; ++cc) {
            const int c = bc + cc * 4;
            const float4 f = *(const float4*)&Bw[(size_t)(k0 + br) * N + col0 + c];
            *(unsigned int*)&Bs[buf][br][c] =
                (unsigned int)f32_bf16(f.x) | ((unsigned int)f32_bf16(f.y) << 16);
            *(unsigned int*)&Bs[buf][br][c + 2] =
                (unsigned int)f32_bf16(f.z) | ((unsigned int)f32_bf16(f.w) << 16);
        }
    };

    const int ksteps = K / TK;
    stage(0, 0);
    __syncthreads();

    const int kbase = half ? 8 : 0;  // A layout: lane<16 -> K 0..7/16..23; else 8..15/24..31

    for (int ks = 0; ks < ksteps; ++ks) {
        const int cur = ks & 1;
        if (ks + 1 < ksteps) {
            stage(cur ^ 1, (ks + 1) * TK);          // overlap with WMMAs below
            if (ks + 2 < ksteps) {                  // warm L2/WGP$ two steps ahead
                __builtin_prefetch(&A[(size_t)(row0 + ar) * K + (ks + 2) * TK + ac], 0, 1);
                __builtin_prefetch(&Bw[(size_t)((ks + 2) * TK + br) * N + col0 + bc], 0, 1);
            }
        }

        // ---- fragments from buffer `cur` ----
        Frag a[2], b[4];
#pragma unroll
        for (int i = 0; i < 2; ++i) {
            const int m = wm * 32 + i * 16 + nlane;
            a[i].q[0] = *(const uint4*)&As[cur][m][kbase];
            a[i].q[1] = *(const uint4*)&As[cur][m][kbase + 16];
        }
#pragma unroll
        for (int j = 0; j < 4; ++j) {
            const int n0 = wn * 64 + j * 16;        // B layout: lane = N col, elems = K (table order)
#pragma unroll
            for (int s = 0; s < 16; ++s)
                ((unsigned short*)&b[j])[s] = Bs[cur][kbase + (s < 8 ? s : s + 8)][n0 + nlane];
        }
#pragma unroll
        for (int i = 0; i < 2; ++i)
#pragma unroll
            for (int j = 0; j < 4; ++j)
                acc[i][j] = __builtin_amdgcn_wmma_f32_16x16x32_bf16(
                    false, a[i].v, false, b[j].v, (short)0, acc[i][j], false, false);

        __syncthreads();   // readers of `cur` done; writers of `cur^1` done
    }

    // ---- epilogue ----
#pragma unroll
    for (int i = 0; i < 2; ++i) {
#pragma unroll
        for (int j = 0; j < 4; ++j) {
#pragma unroll
            for (int v = 0; v < 8; ++v) {
                const int m = row0 + wm * 32 + i * 16 + v + 8 * half;
                const int n = col0 + wn * 64 + j * 16 + nlane;
                const float val = acc[i][j][v] + bias[n];
                if (mode == 0) {
                    outF[(size_t)m * N + n] = val;
                } else {
                    const int which = n >> 10;         // 0=q 1=k 2=v
                    const int h     = (n & 1023) >> 6; // head
                    const int d     = n & 63;          // dim in head
                    const int bidx  = m >> 11;         // batch
                    const int t     = m & 2047;        // token
                    const int hl    = bidx * 16 + h;
                    const unsigned short bv = f32_bf16(val);
                    if (which == 0)      qbuf[((size_t)hl * 2048 + t) * 64 + d] = bv;
                    else if (which == 1) kbuf[((size_t)hl * 2048 + t) * 64 + d] = bv;
                    else                 vbuf[((size_t)hl * 64 + d) * 2048 + t] = bv; // transposed
                }
            }
        }
    }
}

// ---------------------------------------------------------------------------
// Flash attention: one wave (32 threads) per (head, 16-query block).
// Online softmax, all WMMA; P goes through per-wave LDS for D->A relayout.
// ---------------------------------------------------------------------------
__global__ __launch_bounds__(32)
void attn_wmma(const unsigned short* __restrict__ qbuf,
               const unsigned short* __restrict__ kbuf,
               const unsigned short* __restrict__ vbuf,
               float* __restrict__ yatt)
{
    __shared__ __align__(16) unsigned short p_lds[16][32];

    const int lane = threadIdx.x & 31;
    const int hl   = blockIdx.y;       // b*16 + h
    const int b    = hl >> 4;
    const int h    = hl & 15;
    const int q0   = blockIdx.x * 16;
    const int half  = lane >> 4;
    const int nl    = lane & 15;
    const int kbase = half ? 8 : 0;

    // Q fragments: A 16x32 per hd-chunk (hd = 64 -> 2 chunks)
    Frag aq[2];
    const unsigned short* qrow = &qbuf[((size_t)hl * 2048 + q0 + nl) * 64];
#pragma unroll
    for (int c = 0; c < 2; ++c) {
        aq[c].q[0] = *(const uint4*)&qrow[c * 32 + kbase];
        aq[c].q[1] = *(const uint4*)&qrow[c * 32 + kbase + 16];
    }

    v8f acc[4];
    float mrow[8], lrow[8];
#pragma unroll
    for (int j = 0; j < 4; ++j) acc[j] = zero8();
#pragma unroll
    for (int v = 0; v < 8; ++v) { mrow[v] = -1e30f; lrow[v] = 0.0f; }

    const int kend = q0 + 16;   // exclusive bound on keys (causal)
    for (int kb0 = 0; kb0 < kend; kb0 += 32) {
        // K^T B-fragments: lane = key col, elems = hd (table order, K stored token-major)
        Frag bk[2][2];
#pragma unroll
        for (int c = 0; c < 2; ++c) {
#pragma unroll
            for (int tl = 0; tl < 2; ++tl) {
                const unsigned short* krow = &kbuf[((size_t)hl * 2048 + kb0 + tl * 16 + nl) * 64];
                bk[c][tl].q[0] = *(const uint4*)&krow[c * 32 + kbase];
                bk[c][tl].q[1] = *(const uint4*)&krow[c * 32 + kbase + 16];
            }
        }
        v8f s[2];
#pragma unroll
        for (int tl = 0; tl < 2; ++tl) s[tl] = zero8();
#pragma unroll
        for (int c = 0; c < 2; ++c)
#pragma unroll
            for (int tl = 0; tl < 2; ++tl)
                s[tl] = __builtin_amdgcn_wmma_f32_16x16x32_bf16(
                    false, aq[c].v, false, bk[c][tl].v, (short)0, s[tl], false, false);

        // scale, causal mask, online softmax update
        float alpha[8];
#pragma unroll
        for (int v = 0; v < 8; ++v) {
            const int qr = q0 + v + 8 * half;
            float pm = -1e30f;
#pragma unroll
            for (int tl = 0; tl < 2; ++tl) {
                float sc = s[tl][v] * 0.125f;           // 1/sqrt(64)
                const int key = kb0 + tl * 16 + nl;
                sc = (key <= qr) ? sc : -1e30f;
                s[tl][v] = sc;
                pm = fmaxf(pm, sc);
            }
#pragma unroll
            for (int off = 1; off < 16; off <<= 1)
                pm = fmaxf(pm, __shfl_xor(pm, off, 32));
            const float mnew = fmaxf(mrow[v], pm);
            alpha[v] = __expf(mrow[v] - mnew);
            float ps = 0.0f;
#pragma unroll
            for (int tl = 0; tl < 2; ++tl) {
                const float p = __expf(s[tl][v] - mnew);
                s[tl][v] = p;
                ps += p;
            }
#pragma unroll
            for (int off = 1; off < 16; off <<= 1)
                ps += __shfl_xor(ps, off, 32);
            lrow[v] = lrow[v] * alpha[v] + ps;
            mrow[v] = mnew;
        }
#pragma unroll
        for (int j = 0; j < 4; ++j)
#pragma unroll
            for (int v = 0; v < 8; ++v) acc[j][v] *= alpha[v];

        // P: D layout -> LDS -> A layout (bf16)
#pragma unroll
        for (int tl = 0; tl < 2; ++tl)
#pragma unroll
            for (int v = 0; v < 8; ++v)
                p_lds[v + 8 * half][tl * 16 + nl] = f32_bf16(s[tl][v]);
        __syncthreads();
        Frag ap;
        ap.q[0] = *(const uint4*)&p_lds[nl][kbase];
        ap.q[1] = *(const uint4*)&p_lds[nl][kbase + 16];

        // V B-fragments: lane = hd col, elems = keys (table order, V stored transposed)
        Frag bv[4];
#pragma unroll
        for (int j = 0; j < 4; ++j) {
            const unsigned short* vrow = &vbuf[((size_t)hl * 64 + j * 16 + nl) * 2048 + kb0];
            bv[j].q[0] = *(const uint4*)&vrow[kbase];
            bv[j].q[1] = *(const uint4*)&vrow[kbase + 16];
        }
#pragma unroll
        for (int j = 0; j < 4; ++j)
            acc[j] = __builtin_amdgcn_wmma_f32_16x16x32_bf16(
                false, ap.v, false, bv[j].v, (short)0, acc[j], false, false);
        __syncthreads();
    }

    // write y = acc / l into [B*T, C] f32 (token-major for projection GEMM)
#pragma unroll
    for (int j = 0; j < 4; ++j)
#pragma unroll
        for (int v = 0; v < 8; ++v) {
            const int m = q0 + v + 8 * half;
            const int n = h * 64 + j * 16 + nl;
            yatt[((size_t)(b * 2048 + m)) * 1024 + n] = acc[j][v] / lrow[v];
        }
}

// ---------------------------------------------------------------------------
extern "C" void kernel_launch(void* const* d_in, const int* in_sizes, int n_in,
                              void* d_out, int out_size, void* d_ws, size_t ws_size,
                              hipStream_t stream) {
    const float* x      = (const float*)d_in[0];  // [2,2048,1024]
    const float* w_attn = (const float*)d_in[1];  // [1024,3072]
    const float* b_attn = (const float*)d_in[2];  // [3072]
    const float* w_proj = (const float*)d_in[3];  // [1024,1024]
    const float* b_proj = (const float*)d_in[4];  // [1024]

    const size_t QELEMS = (size_t)2 * 16 * 2048 * 64;  // 4,194,304 per buffer
    unsigned short* qbuf = (unsigned short*)d_ws;
    unsigned short* kbuf = qbuf + QELEMS;
    unsigned short* vbuf = kbuf + QELEMS;
    float*          yatt = (float*)(vbuf + QELEMS);    // [4096,1024] f32
    // total workspace use: 3*8MB + 16MB = 40MB

    // QKV GEMM: [4096,1024] x [1024,3072]
    dim3 g1(3072 / TN, 4096 / TM);
    gemm_bf16_wmma<<<g1, 256, 0, stream>>>(x, w_attn, b_attn,
                                           4096, 3072, 1024, /*mode=*/1,
                                           nullptr, qbuf, kbuf, vbuf);

    // Flash attention: 128 query-blocks x 32 (batch*head)
    dim3 g2(2048 / 16, 2 * 16);
    attn_wmma<<<g2, 32, 0, stream>>>(qbuf, kbuf, vbuf, yatt);

    // Output projection: [4096,1024] x [1024,1024] -> d_out (f32)
    dim3 g3(1024 / TN, 4096 / TM);
    gemm_bf16_wmma<<<g3, 256, 0, stream>>>(yatt, w_proj, b_proj,
                                           4096, 1024, 1024, /*mode=*/0,
                                           (float*)d_out, nullptr, nullptr, nullptr);
}
